// simpleMHA_51488067945029
// MI455X (gfx1250) — matrix operand order repaired and checked by reading, not compiled, not run on hardware
//
#include <hip/hip_runtime.h>
#include <hip/hip_bf16.h>

typedef __bf16 bf16_t;
typedef __attribute__((ext_vector_type(16))) __bf16 v16bf;
typedef __attribute__((ext_vector_type(8)))  __bf16 v8bf;
typedef __attribute__((ext_vector_type(8)))  float  v8f;

#define SEQ   2048
#define EMB   2048
#define NH    16
#define HD    128
#define QKVN  6144

static __device__ __forceinline__ v8f zero8() {
  v8f z;
  #pragma unroll
  for (int i = 0; i < 8; ++i) z[i] = 0.0f;
  return z;
}

static __device__ __forceinline__ v8f wmma_bf16(v16bf a, v16bf b, v8f c) {
  return __builtin_amdgcn_wmma_f32_16x16x32_bf16(false, a, false, b, (short)0, c,
                                                 false, false);
}

// async copy 16B global -> LDS (ASYNCcnt-tracked, no VGPR data path)
static __device__ __forceinline__ void async_ld_b128(unsigned lds_off,
                                                     const void* gaddr) {
  asm volatile("global_load_async_to_lds_b128 %0, %1, off"
               :: "v"(lds_off), "v"(gaddr) : "memory");
}
static __device__ __forceinline__ void wait_async0() {
  asm volatile("s_wait_asynccnt 0" ::: "memory");
}

// ---------------------------------------------------------------------------
// fp32 -> bf16 conversion (grid-stride)
// ---------------------------------------------------------------------------
__global__ void cvt_f32_bf16_kernel(const float* __restrict__ in,
                                    bf16_t* __restrict__ out, long n) {
  long i = (long)blockIdx.x * blockDim.x + threadIdx.x;
  long stride = (long)gridDim.x * blockDim.x;
  for (; i < n; i += stride) out[i] = (bf16_t)in[i];
}

// ---------------------------------------------------------------------------
// C[M,N] = A[M,K](bf16) * W[N,K]^T(bf16) + bias[N]  (fp32 out)
// Block: 256 thr (8 waves), tile 128x128. Wave: 32x64 (8 WMMA / K-step).
// BOTH A (128x32) and B (128x32) tiles staged in LDS per K-step via
// global_load_async_to_lds_b128, double-buffered; all WMMA operands come
// from conflict-free ds_load_b128 (80B row stride). Grid: (N/128, M/128).
// ---------------------------------------------------------------------------
#define KSTEP 32
#define BLK_M 128
#define LSTRIDE 40   // bf16 elems; 80B row stride -> conflict-free ds_load_b128

__global__ __launch_bounds__(256) void gemm_bf16_wmma_kernel(
    const bf16_t* __restrict__ A, const bf16_t* __restrict__ W,
    const float* __restrict__ bias, float* __restrict__ C,
    int M, int N, int K) {
  __shared__ __attribute__((aligned(128))) bf16_t As[2][BLK_M][LSTRIDE];
  __shared__ __attribute__((aligned(128))) bf16_t Bs[2][BLK_M][LSTRIDE];

  const int tid  = threadIdx.x;
  const int lane = tid & 31;
  const int wid  = tid >> 5;
  const int ln   = lane & 15;
  const int hf   = lane >> 4;
  const int wm   = wid >> 1;          // 0..3 : 32-row band
  const int wn   = wid & 1;           // 0..1 : 64-col band
  const int m_blk = blockIdx.y * BLK_M;
  const int n_blk = blockIdx.x * 128;
  const int n0    = n_blk + wn * 64;

  v8f acc[2][4];
  #pragma unroll
  for (int s = 0; s < 2; ++s)
    #pragma unroll
    for (int t = 0; t < 4; ++t) acc[s][t] = zero8();

  // stage A+B tiles for K-step kb into buffer kb&1 (1024 x 16B, 4/thread)
  auto stage = [&](int kb) {
    const int buf = kb & 1;
    const int k0  = kb * KSTEP;
    #pragma unroll
    for (int j = 0; j < 2; ++j) {
      const int c   = tid + 256 * j;
      const int row = c >> 2;        // 4x16B chunks per 32-elem row
      const int kp  = c & 3;
      async_ld_b128((unsigned)(size_t)&As[buf][row][kp * 8],
                    A + (size_t)(m_blk + row) * K + k0 + kp * 8);
      async_ld_b128((unsigned)(size_t)&Bs[buf][row][kp * 8],
                    W + (size_t)(n_blk + row) * K + k0 + kp * 8);
    }
  };

  stage(0);
  wait_async0();
  __syncthreads();

  const int nkb = K / KSTEP;
  for (int kb = 0; kb < nkb; ++kb) {
    if (kb + 1 < nkb) stage(kb + 1);   // overlap next tile's async copies

    const int buf = kb & 1;
    union F16 { v16bf v; v8bf h[2]; };
    F16 af[2];
    #pragma unroll
    for (int s = 0; s < 2; ++s) {
      const int row = wm * 32 + s * 16 + ln;
      af[s].h[0] = *(const v8bf*)(&As[buf][row][16 * hf]);       // K 0-7/16-23
      af[s].h[1] = *(const v8bf*)(&As[buf][row][16 * hf + 8]);   // K 8-15/24-31
    }
    #pragma unroll
    for (int t = 0; t < 4; ++t) {
      const int row = wn * 64 + t * 16 + ln;
      F16 bfv;                                  // 16 contiguous K, 2x16B
      bfv.h[0] = *(const v8bf*)(&Bs[buf][row][16 * hf]);
      bfv.h[1] = *(const v8bf*)(&Bs[buf][row][16 * hf + 8]);
      acc[0][t] = wmma_bf16(af[0].v, bfv.v, acc[0][t]);
      acc[1][t] = wmma_bf16(af[1].v, bfv.v, acc[1][t]);
    }

    wait_async0();       // own async stores to LDS done
    __syncthreads();     // everyone's done -> next buffer valid / reusable
  }

  #pragma unroll
  for (int t = 0; t < 4; ++t) {
    const int n = n0 + t * 16 + ln;
    const float b = bias[n];
    #pragma unroll
    for (int s = 0; s < 2; ++s) {
      #pragma unroll
      for (int r = 0; r < 8; ++r) {
        const int m = m_blk + wm * 32 + s * 16 + 8 * hf + r;
        C[(size_t)m * N + n] = acc[s][t][r] + b;
      }
    }
  }
}

// ---------------------------------------------------------------------------
// RoPE + repack: qkv_f32[S,6144] -> Q[H,S,D] (scaled 1/sqrt(D)), K[H,S,D],
// V^T[H,D,S] all bf16.  grid (S, H), block 128 (=D)
// ---------------------------------------------------------------------------
__global__ __launch_bounds__(128) void rope_pack_kernel(
    const float* __restrict__ qkv, bf16_t* __restrict__ Qo,
    bf16_t* __restrict__ Ko, bf16_t* __restrict__ VTo) {
  const int s = blockIdx.x;
  const int h = blockIdx.y;
  const int d = threadIdx.x;
  const float* row = qkv + (size_t)s * QKVN;
  const int qoff = h * HD + d;
  const int koff = EMB + h * HD + d;
  float q = row[qoff];
  float k = row[koff];
  float v = row[2 * EMB + h * HD + d];
  float qr = q, kr = k;
  if (d < 64) {
    const int i = d & 31;
    const float freq = (float)s * __powf(10000.0f, -(float)i / 32.0f);
    float sn, cs;
    __sincosf(freq, &sn, &cs);
    if (d < 32) {
      const float q2 = row[qoff + 32];
      const float k2 = row[koff + 32];
      qr = q * cs - q2 * sn;
      kr = k * cs - k2 * sn;
    } else {
      const float q1 = row[qoff - 32];
      const float k1 = row[koff - 32];
      qr = q1 * sn + q * cs;
      kr = k1 * sn + k * cs;
    }
  }
  const float qscale = 0.08838834764831845f;    // 1/sqrt(128)
  Qo[((size_t)h * SEQ + s) * HD + d]  = (bf16_t)(qr * qscale);
  Ko[((size_t)h * SEQ + s) * HD + d]  = (bf16_t)kr;
  VTo[((size_t)h * HD + d) * SEQ + s] = (bf16_t)v;
}

// ---------------------------------------------------------------------------
// Causal flash attention, one wave per 16-query tile per head.
// Q,K: [H,S,D] bf16 (Q pre-scaled); Vt: [H,D,S] bf16; O: [S, H*D] bf16
// grid: (S/128, H), block 256 (8 waves). No block barriers (divergent
// causal trip counts); P transpose via per-wave LDS slab + s_wait_dscnt.
// ---------------------------------------------------------------------------
__global__ __launch_bounds__(256) void flash_attn_wmma_kernel(
    const bf16_t* __restrict__ Q, const bf16_t* __restrict__ Kc,
    const bf16_t* __restrict__ Vt, bf16_t* __restrict__ O) {
  __shared__ __attribute__((aligned(64))) bf16_t plds[8][16][32];

  const int lane = threadIdx.x & 31;
  const int wid  = threadIdx.x >> 5;
  const int ln   = lane & 15;
  const int hf   = lane >> 4;
  const int head = blockIdx.y;
  const int q0   = (blockIdx.x * 8 + wid) * 16;

  union F16 { v16bf v; v8bf h[2]; };

  F16 qa[4];
  const bf16_t* qbase = Q + ((size_t)head * SEQ + q0 + ln) * HD;
  #pragma unroll
  for (int c = 0; c < 4; ++c) {
    qa[c].h[0] = *(const v8bf*)(qbase + 32 * c + 16 * hf);
    qa[c].h[1] = *(const v8bf*)(qbase + 32 * c + 16 * hf + 8);
  }

  float mrow[8], lrow[8];
  v8f o[8];
  #pragma unroll
  for (int r = 0; r < 8; ++r) { mrow[r] = -3.0e38f; lrow[r] = 0.0f; }
  #pragma unroll
  for (int d = 0; d < 8; ++d) o[d] = zero8();

  const int nkb = (q0 + 16 + 31) >> 5;
  for (int kb = 0; kb < nkb; ++kb) {
    const int kbase = kb * 32;

    v8f s0 = zero8(), s1 = zero8();
    const bf16_t* kp0 = Kc + ((size_t)head * SEQ + kbase + ln) * HD + 16 * hf;
    const bf16_t* kp1 = kp0 + (size_t)16 * HD;
    #pragma unroll
    for (int c = 0; c < 4; ++c) {
      v16bf f0 = *(const v16bf*)(kp0 + 32 * c);
      v16bf f1 = *(const v16bf*)(kp1 + 32 * c);
      s0 = wmma_bf16(qa[c].v, f0, s0);
      s1 = wmma_bf16(qa[c].v, f1, s1);
    }

    float p0[8], p1[8];
    const int mbase = q0 + 8 * hf;
    #pragma unroll
    for (int r = 0; r < 8; ++r) {
      const int m = mbase + r;
      p0[r] = (kbase + ln      > m) ? -3.0e38f : s0[r];
      p1[r] = (kbase + 16 + ln > m) ? -3.0e38f : s1[r];
    }

    float t[8];
    #pragma unroll
    for (int r = 0; r < 8; ++r) t[r] = fmaxf(p0[r], p1[r]);
    #pragma unroll
    for (int off = 1; off < 16; off <<= 1) {
      #pragma unroll
      for (int r = 0; r < 8; ++r)
        t[r] = fmaxf(t[r], __shfl_xor(t[r], off, 32));
    }

    float rs[8];
    #pragma unroll
    for (int r = 0; r < 8; ++r) {
      const float mn = fmaxf(mrow[r], t[r]);
      const float sc = __expf(mrow[r] - mn);
      const float e0 = __expf(p0[r] - mn);
      const float e1 = __expf(p1[r] - mn);
      mrow[r] = mn;
      lrow[r] *= sc;
      rs[r] = e0 + e1;
      p0[r] = e0;
      p1[r] = e1;
      #pragma unroll
      for (int d = 0; d < 8; ++d) o[d][r] *= sc;
    }
    #pragma unroll
    for (int off = 1; off < 16; off <<= 1) {
      #pragma unroll
      for (int r = 0; r < 8; ++r) rs[r] += __shfl_xor(rs[r], off, 32);
    }
    #pragma unroll
    for (int r = 0; r < 8; ++r) lrow[r] += rs[r];

    #pragma unroll
    for (int r = 0; r < 8; ++r) {
      plds[wid][8 * hf + r][ln]      = (bf16_t)p0[r];
      plds[wid][8 * hf + r][16 + ln] = (bf16_t)p1[r];
    }
    asm volatile("s_wait_dscnt 0" ::: "memory");

    F16 pf;
    pf.h[0] = *(const v8bf*)(&plds[wid][ln][16 * hf]);
    pf.h[1] = *(const v8bf*)(&plds[wid][ln][16 * hf + 8]);

    const bf16_t* vb = Vt + ((size_t)head * HD + ln) * SEQ + kbase + 16 * hf;
    #pragma unroll
    for (int d = 0; d < 8; ++d) {
      v16bf vf = *(const v16bf*)(vb + (size_t)d * 16 * SEQ);
      o[d] = wmma_bf16(pf.v, vf, o[d]);
    }
  }

  float inv[8];
  #pragma unroll
  for (int r = 0; r < 8; ++r) inv[r] = 1.0f / lrow[r];
  #pragma unroll
  for (int d = 0; d < 8; ++d) {
    #pragma unroll
    for (int r = 0; r < 8; ++r) {
      const int srow = q0 + 8 * hf + r;
      O[(size_t)srow * EMB + head * HD + d * 16 + ln] = (bf16_t)(o[d][r] * inv[r]);
    }
  }
}

// ---------------------------------------------------------------------------
// launcher
// ---------------------------------------------------------------------------
extern "C" void kernel_launch(void* const* d_in, const int* in_sizes, int n_in,
                              void* d_out, int out_size, void* d_ws, size_t ws_size,
                              hipStream_t stream) {
  const float* x      = (const float*)d_in[0];
  const float* Wqkv   = (const float*)d_in[1];
  const float* Wqkv_b = (const float*)d_in[2];
  const float* Wo     = (const float*)d_in[3];
  const float* Wo_b   = (const float*)d_in[4];
  float* out = (float*)d_out;
  char* ws = (char*)d_ws;

  const size_t SZ_XBF   = (size_t)SEQ * EMB * 2;        //  8 MB
  const size_t SZ_WQKV  = (size_t)QKVN * EMB * 2;       // 24 MB
  const size_t OFF_XBF  = 0;
  const size_t OFF_WQKV = SZ_XBF;
  const size_t OFF_WO   = OFF_WQKV + SZ_WQKV;
  const size_t OFF_QKVF = OFF_WO + SZ_XBF;
  const size_t OFF_Q = 0;                 // aliases xbf (dead after GEMM1)
  const size_t OFF_K = SZ_XBF;            // aliases wqkvbf
  const size_t OFF_V = 2 * SZ_XBF;
  const size_t OFF_ATTN = OFF_QKVF;       // aliases qkv_f32 (dead after rope)

  bf16_t* xbf    = (bf16_t*)(ws + OFF_XBF);
  bf16_t* wqkvbf = (bf16_t*)(ws + OFF_WQKV);
  bf16_t* wobf   = (bf16_t*)(ws + OFF_WO);
  float*  qkvf   = (float*)(ws + OFF_QKVF);
  bf16_t* qbf    = (bf16_t*)(ws + OFF_Q);
  bf16_t* kbf    = (bf16_t*)(ws + OFF_K);
  bf16_t* vtbf   = (bf16_t*)(ws + OFF_V);
  bf16_t* attnbf = (bf16_t*)(ws + OFF_ATTN);

  cvt_f32_bf16_kernel<<<4096, 256, 0, stream>>>(x, xbf, (long)SEQ * EMB);
  cvt_f32_bf16_kernel<<<4096, 256, 0, stream>>>(Wqkv, wqkvbf, (long)QKVN * EMB);
  cvt_f32_bf16_kernel<<<4096, 256, 0, stream>>>(Wo, wobf, (long)EMB * EMB);

  gemm_bf16_wmma_kernel<<<dim3(QKVN / 128, SEQ / 128), 256, 0, stream>>>(
      xbf, wqkvbf, Wqkv_b, qkvf, SEQ, QKVN, EMB);

  rope_pack_kernel<<<dim3(SEQ, NH), HD, 0, stream>>>(qkvf, qbf, kbf, vtbf);

  flash_attn_wmma_kernel<<<dim3(SEQ / 128, NH), 256, 0, stream>>>(
      qbf, kbf, vtbf, attnbf);

  gemm_bf16_wmma_kernel<<<dim3(EMB / 128, SEQ / 128), 256, 0, stream>>>(
      attnbf, wobf, Wo_b, out, SEQ, EMB, EMB);
}
